// GraphAttentionLayer_7421703487703
// MI455X (gfx1250) — matrix operand order repaired and checked by reading, not compiled, not run on hardware
//
#include <hip/hip_runtime.h>
#include <hip/hip_bf16.h>

// MI455X / gfx1250, wave32. Flash-style fused GAT:
//   HBM traffic ~ adj (134MB, read once) + x (17MB) + out (4MB) => ~6.7us at 23.3TB/s.
//   h kept as f16 in workspace (2MB) -> L2-resident across its N/16-fold reuse.
//   All matmuls via v_wmma_f32_16x16x32_f16.

typedef __attribute__((ext_vector_type(16))) _Float16 v16h;
typedef __attribute__((ext_vector_type(8)))  float    v8f;
typedef __attribute__((ext_vector_type(8)))  float    f32x8;
typedef __attribute__((ext_vector_type(8)))  int      i32x8;

#define ALPHA 0.2f
#define B_   8
#define N_   2048
#define FIN  256
#define FOUT 64

// ---------------------------------------------------------------------------
// Kernel 0: W (256x64 f32, row-major) -> W_t (64x256 f16, col-major / B-friendly)
// ---------------------------------------------------------------------------
__global__ __launch_bounds__(256) void k_wt(const float* __restrict__ W,
                                            _Float16* __restrict__ Wt) {
  int i = blockIdx.x * 256 + threadIdx.x;        // 0..16383
  int col = i >> 8;                              // /256
  int k   = i & 255;
  Wt[col * FIN + k] = (_Float16)W[k * FOUT + col];
}

// ---------------------------------------------------------------------------
// Kernel 1: h = x @ W via WMMA, stored transposed as h_t[b][col][m] (f16).
// One wave per 16-row tile. C tile transposed through LDS so h_t stores are
// contiguous 32B per lane.
// ---------------------------------------------------------------------------
__global__ __launch_bounds__(256) void k_h(const float* __restrict__ x,
                                           const _Float16* __restrict__ Wt,
                                           _Float16* __restrict__ ht) {
  __shared__ float lds[8][FOUT * 17];            // per-wave 64 cols x 16 rows (+pad)
  const int lane = threadIdx.x & 31;
  const int wv   = threadIdx.x >> 5;
  const int gw   = blockIdx.x * 8 + wv;          // 0..1023 row tiles
  const int row0 = gw * 16;                      // global row (B*N space)
  const int b    = row0 / N_;
  const int r0b  = row0 % N_;
  const int half = lane >> 4;                    // 0 or 1
  const int l15  = lane & 15;

  v8f acc[4] = {};
  const float* xrow = x + (size_t)(row0 + l15) * FIN;

  for (int k0 = 0; k0 < FIN; k0 += 32) {
    // A operand 16x32 f16: lane<16 holds K 0..15, lane>=16 K 16..31
    f32x8 xa = *(const f32x8*)(xrow + k0 + half * 16);
    f32x8 xb = *(const f32x8*)(xrow + k0 + half * 16 + 8);
    v16h A;
#pragma unroll
    for (int i = 0; i < 8; i++) { A[i] = (_Float16)xa[i]; A[8 + i] = (_Float16)xb[i]; }
#pragma unroll
    for (int t = 0; t < 4; t++) {
      // B operand 32x16 f16: lane<16 -> K 0..15, lane>=16 -> K 16..31, col = lane&15
      v16h Bv = *(const v16h*)(Wt + (size_t)(16 * t + l15) * FIN + k0 + half * 16);
      acc[t] = __builtin_amdgcn_wmma_f32_16x16x32_f16(false, A, false, Bv,
                                                      (short)0, acc[t], false, false);
    }
  }

  // C layout: VGPR r -> row r + half*8, col = 16*t + l15. Transpose via LDS.
#pragma unroll
  for (int t = 0; t < 4; t++)
#pragma unroll
    for (int r = 0; r < 8; r++)
      lds[wv][(16 * t + l15) * 17 + r + half * 8] = acc[t][r];

  // Wave-local LDS dependency; compiler inserts s_wait_dscnt.
#pragma unroll
  for (int cc = 0; cc < 2; cc++) {
    int col = lane * 2 + cc;
    v16h o;
#pragma unroll
    for (int j = 0; j < 16; j++) o[j] = (_Float16)lds[wv][col * 17 + j];
    *(v16h*)(ht + (size_t)b * FOUT * N_ + (size_t)col * N_ + r0b) = o;
  }
}

// ---------------------------------------------------------------------------
// Kernel 2: e_l[b,m] = h[b,m,:]@a_left ; e_r[b,m] = h[b,m,:]@a_right
// h_t layout makes these loads coalesced across threads (consecutive m).
// ---------------------------------------------------------------------------
__global__ __launch_bounds__(256) void k_e(const _Float16* __restrict__ ht,
                                           const float* __restrict__ a,
                                           float* __restrict__ el,
                                           float* __restrict__ er) {
  int i = blockIdx.x * 256 + threadIdx.x;        // 0..16383
  int b = i / N_, m = i % N_;
  const _Float16* hb = ht + (size_t)b * FOUT * N_ + m;
  float sl = 0.f, sr = 0.f;
#pragma unroll 8
  for (int c = 0; c < FOUT; c++) {
    float v = (float)hb[(size_t)c * N_];
    sl += v * a[c];
    sr += v * a[FOUT + c];
  }
  el[i] = sl;
  er[i] = sr;
}

// ---------------------------------------------------------------------------
// Kernel 2b: per-batch max of e_r -> enables single-pass stable softmax.
// ---------------------------------------------------------------------------
__global__ __launch_bounds__(256) void k_max(const float* __restrict__ er,
                                             float* __restrict__ maxer) {
  __shared__ float red[256];
  int b = blockIdx.x;
  float m = -1e30f;
  for (int i = threadIdx.x; i < N_; i += 256) m = fmaxf(m, er[b * N_ + i]);
  red[threadIdx.x] = m;
  __syncthreads();
  for (int s = 128; s > 0; s >>= 1) {
    if (threadIdx.x < (unsigned)s) red[threadIdx.x] = fmaxf(red[threadIdx.x], red[threadIdx.x + s]);
    __syncthreads();
  }
  if (threadIdx.x == 0) maxer[b] = red[0];
}

// ---------------------------------------------------------------------------
// Kernel 3: fused masked-softmax + attn@h + ELU. One wave per 16 output rows,
// streaming m in chunks of 32. C_row = lrelu(e_l[i] + max_m e_r) >= max_m e[i,m],
// so exp(e - C_row) <= 1: no online rescaling needed, adj read exactly once.
// ---------------------------------------------------------------------------
__global__ __launch_bounds__(256) void k_attn(const int* __restrict__ adj,
                                              const _Float16* __restrict__ ht,
                                              const float* __restrict__ el,
                                              const float* __restrict__ er,
                                              const float* __restrict__ maxer,
                                              float* __restrict__ out) {
  const int lane = threadIdx.x & 31;
  const int wv   = threadIdx.x >> 5;
  const int gw   = blockIdx.x * 8 + wv;          // 0..1023
  const int b    = gw >> 7;                      // /128
  const int r0b  = (gw & 127) * 16;
  const int half = lane >> 4;
  const int l15  = lane & 15;
  const int rb1  = half * 16;                    // A-operand K-run base
  const int koffB = half * 16;                   // B-operand K base

  const float elrow = el[b * N_ + r0b + l15];
  float tc = elrow + maxer[b];
  const float crow = tc > 0.f ? tc : ALPHA * tc; // per-row upper bound on e

  const int*      adjrow = adj + (size_t)(b * N_ + r0b + l15) * N_;
  const float*    erb    = er + b * N_;
  const _Float16* hb     = ht + (size_t)b * FOUT * N_;

  v8f acc[4] = {};
  float lsum = 0.f;

  for (int m0 = 0; m0 < N_; m0 += 32) {
    __builtin_prefetch(adjrow + m0 + 64, 0, 1);  // stream adj ahead (global_prefetch_b8)
    i32x8 ad0 = *(const i32x8*)(adjrow + m0 + rb1);
    i32x8 ad1 = *(const i32x8*)(adjrow + m0 + rb1 + 8);
    f32x8 er0 = *(const f32x8*)(erb + m0 + rb1);
    f32x8 er1 = *(const f32x8*)(erb + m0 + rb1 + 8);

    v16h A;  // masked exp(e - crow), laid out directly as the WMMA A operand
#pragma unroll
    for (int i = 0; i < 8; i++) {
      float t0 = elrow + er0[i]; t0 = t0 > 0.f ? t0 : ALPHA * t0;
      float p0 = (ad0[i] > 0) ? __expf(t0 - crow) : 0.f;
      lsum += p0; A[i] = (_Float16)p0;
      float t1 = elrow + er1[i]; t1 = t1 > 0.f ? t1 : ALPHA * t1;
      float p1 = (ad1[i] > 0) ? __expf(t1 - crow) : 0.f;
      lsum += p1; A[8 + i] = (_Float16)p1;
    }
#pragma unroll
    for (int t = 0; t < 4; t++) {
      v16h Bv = *(const v16h*)(hb + (size_t)(16 * t + l15) * N_ + m0 + koffB);
      acc[t] = __builtin_amdgcn_wmma_f32_16x16x32_f16(false, A, false, Bv,
                                                      (short)0, acc[t], false, false);
    }
  }

  // Row denominator: lanes L and L+16 each hold half the row's p-sum.
  float dfull = lsum + __shfl_xor(lsum, 16, 32);

  float* ob = out + (size_t)(b * N_ + r0b) * FOUT;
#pragma unroll
  for (int t = 0; t < 4; t++)
#pragma unroll
    for (int r = 0; r < 8; r++) {
      int rowl = r + half * 8;                   // C-layout row of this VGPR
      float dn = __shfl(dfull, rowl, 32);        // lane rowl holds that row's denom
      float v  = acc[t][r] / dn;
      v = v > 0.f ? v : (__expf(v) - 1.f);       // ELU
      ob[(size_t)rowl * FOUT + 16 * t + l15] = v;
    }
}

// ---------------------------------------------------------------------------
extern "C" void kernel_launch(void* const* d_in, const int* in_sizes, int n_in,
                              void* d_out, int out_size, void* d_ws, size_t ws_size,
                              hipStream_t stream) {
  (void)in_sizes; (void)n_in; (void)out_size; (void)ws_size;
  const float* x   = (const float*)d_in[0];   // (8,2048,256) f32
  const int*   adj = (const int*)d_in[1];     // (8,2048,2048) i32
  const float* W   = (const float*)d_in[2];   // (256,64) f32
  const float* a   = (const float*)d_in[3];   // (128,1) f32
  float* out = (float*)d_out;                 // (8,2048,64) f32

  char* ws = (char*)d_ws;
  _Float16* Wt = (_Float16*)(ws);                               // 32 KB
  _Float16* ht = (_Float16*)(ws + 32768);                       // 2 MB
  float*    el = (float*)(ws + 32768 + 2097152);                // 64 KB
  float*    er = (float*)(ws + 32768 + 2097152 + 65536);        // 64 KB
  float*    mx = (float*)(ws + 32768 + 2097152 + 131072);       // 32 B

  hipLaunchKernelGGL(k_wt,   dim3(64),  dim3(256), 0, stream, W, Wt);
  hipLaunchKernelGGL(k_h,    dim3(128), dim3(256), 0, stream, x, Wt, ht);
  hipLaunchKernelGGL(k_e,    dim3(64),  dim3(256), 0, stream, ht, a, el, er);
  hipLaunchKernelGGL(k_max,  dim3(8),   dim3(256), 0, stream, er, mx);
  hipLaunchKernelGGL(k_attn, dim3(128), dim3(256), 0, stream, adj, ht, el, er, mx, out);
}
